// seg_head_Deform_62311385531079
// MI455X (gfx1250) — matrix operand order repaired and checked by reading, not compiled, not run on hardware
//
#include <hip/hip_runtime.h>

// ---------------------------------------------------------------- constants
#define Bn      8
#define Tn      3
#define Cn      256
#define HIDn    256
#define Mh      8
#define Pn      4
#define NLVL    3
#define NLAY    6
#define FFNn    1024
#define LEN     5376            // 64*64 + 32*32 + 16*16
#define BLr     43008           // Bn * LEN

typedef __attribute__((ext_vector_type(16))) __bf16 v16bf;
typedef __attribute__((ext_vector_type(8)))  float  v8f;

__device__ __forceinline__ unsigned short f2bf_bits(float f) {
    union { float f; unsigned u; } v; v.f = f;
    unsigned u = v.u;
    unsigned r = u + 0x7FFFu + ((u >> 16) & 1u);   // round-to-nearest-even
    return (unsigned short)(r >> 16);
}
__device__ __forceinline__ __bf16 us2bf(unsigned short u) {
    return __builtin_bit_cast(__bf16, u);
}
__device__ __forceinline__ void lvl_of(int qi, int& l, int& side, int& lvlOff) {
    if (qi < 4096)      { l = 0; side = 64; lvlOff = 0; }
    else if (qi < 5120) { l = 1; side = 32; lvlOff = 4096; }
    else                { l = 2; side = 16; lvlOff = 5120; }
}

// ---------------------------------------------------------------- WMMA GEMM
// C[row,col] = sum_k A[row,k] * Bmat[k,col] (+bias, optional ReLU)
// A: bf16 row-major [M,K].  Bt: bf16 "column-major" [N,K] (Bt[n*K+k]=Bmat[k,n]).
// Block = 4 waves (32x4). All 4 waves share one M-tile: the 16x32 A chunk is
// staged into LDS ONCE per block by wave 0 using gfx1250 async LDS loads
// (GLOBAL_LOAD_ASYNC_TO_LDS_B128, ASYNCcnt), double-buffered so chunk k+1
// streams in while chunk k is consumed via ds_load + v_wmma.
// M must be a multiple of 16 (true for every call site here).
__global__ __launch_bounds__(128)
void gemm_bf16(const unsigned short* __restrict__ A,
               const unsigned short* __restrict__ Bt,
               const float* __restrict__ bias,
               float* __restrict__ Cf, unsigned short* __restrict__ Cb,
               int Mrows, int N, int K,
               long strideAz, long strideCz, int relu)
{
    __shared__ unsigned short Atile[2][16 * 32];     // 2 x 1KB double buffer

    // Escape the LDS tile address into asm once: the async ops below write it
    // behind the compiler's back, this keeps subsequent ds reads live/ordered.
    {
        unsigned short* esc = &Atile[0][0];
        asm volatile("" :: "v"(esc) : "memory");
    }

    const int lane  = threadIdx.x;          // 0..31
    const int wave  = threadIdx.y;          // 0..3
    const int mTile = blockIdx.x;
    const int nTiles = N >> 4;
    int nTile = blockIdx.y * 4 + wave;
    const bool active = (nTile < nTiles);
    if (!active) nTile = nTiles - 1;        // keep wave alive for barriers; no store
    const int lr = lane & 15;
    const int hi = lane >> 4;

    const unsigned short* Abase =
        A + (long)blockIdx.z * strideAz + (long)mTile * 16 * K;
    // B fragment: lane(hi,lr) holds column (nTile*16+lr), K = k0 + hi*16 + e
    const unsigned short* bcol = Bt + (long)(nTile * 16 + lr) * K + hi * 16;

    const int nCh = K >> 5;                 // # of 32-wide K chunks
    const int r = lane >> 2;                // async copy: 8 rows x 4 x 16B segs
    const int s = lane & 3;

    // issue chunk 0 into buffer 0
    if (wave == 0) {
        const unsigned short* g0 = Abase + (long)r * K + s * 8;
        unsigned l0 = (unsigned)(size_t)&Atile[0][r * 32 + s * 8];
        asm volatile("global_load_async_to_lds_b128 %0, %1, off"
                     :: "v"(l0), "v"(g0) : "memory");
        const unsigned short* g1 = g0 + 8L * K;
        unsigned l1 = (unsigned)(size_t)&Atile[0][(r + 8) * 32 + s * 8];
        asm volatile("global_load_async_to_lds_b128 %0, %1, off"
                     :: "v"(l1), "v"(g1) : "memory");
    }

    v8f acc = {};
    for (int kc = 0; kc < nCh; ++kc) {
        if (wave == 0)
            asm volatile("s_wait_asynccnt 0" ::: "memory");  // chunk kc landed
        __syncthreads();                                     // visible to all waves
        if (wave == 0 && (kc + 1) < nCh) {                   // prefetch chunk kc+1
            const int k1 = (kc + 1) << 5;
            unsigned short* dst = &Atile[(kc + 1) & 1][0];
            const unsigned short* g0 = Abase + (long)r * K + k1 + s * 8;
            unsigned l0 = (unsigned)(size_t)&dst[r * 32 + s * 8];
            asm volatile("global_load_async_to_lds_b128 %0, %1, off"
                         :: "v"(l0), "v"(g0) : "memory");
            const unsigned short* g1 = g0 + 8L * K;
            unsigned l1 = (unsigned)(size_t)&dst[(r + 8) * 32 + s * 8];
            asm volatile("global_load_async_to_lds_b128 %0, %1, off"
                         :: "v"(l1), "v"(g1) : "memory");
        }
        // A fragment from LDS in this code's own k order (not the ISA table order):
        // lane(hi,lr): K=hi*16+e (contiguous, matches B fragment)
        const unsigned short* As = &Atile[kc & 1][0];
        v16bf a, b;
#pragma unroll
        for (int e = 0; e < 8; ++e) {
            a[e]     = us2bf(As[lr * 32 + hi * 16 + e]);
            a[8 + e] = us2bf(As[lr * 32 + hi * 16 + 8 + e]);
        }
        const int k0 = kc << 5;
#pragma unroll
        for (int e = 0; e < 16; ++e)
            b[e] = us2bf(bcol[k0 + e]);
        acc = __builtin_amdgcn_wmma_f32_16x16x32_bf16(
                  false, a, false, b, (short)0, acc, false, false);
    }

    if (!active) return;
    const int col = nTile * 16 + lr;
    const float bv = bias ? bias[col] : 0.f;
    const long cz = (long)blockIdx.z * strideCz;
#pragma unroll
    for (int j = 0; j < 8; ++j) {
        const int row = mTile * 16 + hi * 8 + j;     // C/D layout: VGPR j, lane half
        float v = acc[j] + bv;
        if (relu) v = fmaxf(v, 0.f);
        const long idx = cz + (long)row * N + col;
        if (Cf) Cf[idx] = v;
        if (Cb) Cb[idx] = f2bf_bits(v);
    }
}

// ---------------------------------------------------------------- weight prep
__global__ __launch_bounds__(256)
void k_w_transpose(const float* __restrict__ in, unsigned short* __restrict__ out,
                   int K, int N, long total)
{
    long i = (long)blockIdx.x * 256 + threadIdx.x;
    if (i >= total) return;
    long kn  = (long)K * N;
    long mat = i / kn;
    long r   = i - mat * kn;
    int  n   = (int)(r / K);
    int  k   = (int)(r - (long)n * K);
    out[i] = f2bf_bits(in[mat * kn + (long)k * N + n]);
}
__global__ __launch_bounds__(256)
void k_w_convert(const float* __restrict__ in, unsigned short* __restrict__ out, long total)
{
    long i = (long)blockIdx.x * 256 + threadIdx.x;
    if (i >= total) return;
    out[i] = f2bf_bits(in[i]);
}

// ---------------------------------------------------------------- pos embed
__global__ __launch_bounds__(256)
void k_pos(const float* __restrict__ level_embed, float* __restrict__ POS)
{
    int gid = blockIdx.x * 256 + threadIdx.x;
    if (gid >= LEN * 256) return;
    int c = gid & 255, qi = gid >> 8;
    int l, side, lvlOff; lvl_of(qi, l, side, lvlOff);
    int pix = qi - lvlOff, y = pix / side, x = pix % side;
    const float scale = 6.283185307179586f;
    int j; float v;
    if (c < 128) { j = c;       v = (float)(y + 1) / ((float)side + 1e-6f) * scale; }
    else         { j = c - 128; v = (float)(x + 1) / ((float)side + 1e-6f) * scale; }
    float dim = powf(10000.f, (2.f * (float)(j >> 1)) / 128.f);
    float arg = v / dim;
    float e = (j & 1) ? cosf(arg) : sinf(arg);
    POS[gid] = e + level_embed[l * 256 + c];
}

// ---------------------------------------------------------------- time fusion
__global__ __launch_bounds__(256)
void k_squeeze(const float* __restrict__ f0, const float* __restrict__ f1,
               const float* __restrict__ f2_, float* __restrict__ SQ)
{
    __shared__ float s1[256];
    int blk = blockIdx.x;               // (b*3+l)*768 + tc
    int tc = blk % 768, bl = blk / 768, l = bl % 3, b = bl / 3;
    const float* f = (l == 0) ? f0 : (l == 1) ? f1 : f2_;
    int side = (l == 0) ? 64 : (l == 1) ? 32 : 16;
    int HW = side * side;
    const float* p = f + ((long)b * 768 + tc) * HW;
    float s = 0.f;
    for (int i = threadIdx.x; i < HW; i += 256) s += p[i];
    s1[threadIdx.x] = s; __syncthreads();
    for (int st = 128; st > 0; st >>= 1) {
        if (threadIdx.x < st) s1[threadIdx.x] += s1[threadIdx.x + st];
        __syncthreads();
    }
    if (threadIdx.x == 0) SQ[blk] = s1[0] / (float)HW;
}

__global__ __launch_bounds__(256)
void k_tfl(const float* __restrict__ SQ,
           const float* __restrict__ fc1w, const float* __restrict__ fc1b,
           const float* __restrict__ fc2w, const float* __restrict__ fc2b,
           float* __restrict__ F2)
{
    __shared__ float f1s[256];
    int blk = blockIdx.x, l = blk % 3, b = blk / 3, t = threadIdx.x;
    const float* sq = SQ + (b * 3 + l) * 768;
    const float* w1 = fc1w + (long)l * 768 * 256;
    float a = fc1b[l * 256 + t];
    for (int k = 0; k < 768; ++k) a += sq[k] * w1[(long)k * 256 + t];
    f1s[t] = fmaxf(a, 0.f);
    __syncthreads();
    const float* w2 = fc2w + (long)l * 256 * 768;
    float lg[3];
    for (int tt = 0; tt < 3; ++tt) {
        float s = fc2b[l * 768 + tt * 256 + t];
        for (int k = 0; k < 256; ++k) s += f1s[k] * w2[(long)k * 768 + tt * 256 + t];
        lg[tt] = s;
    }
    float mx = fmaxf(lg[0], fmaxf(lg[1], lg[2]));
    float e0 = expf(lg[0] - mx), e1 = expf(lg[1] - mx), e2 = expf(lg[2] - mx);
    float inv = 1.f / (e0 + e1 + e2);
    float* out = F2 + ((long)(b * 3 + l) * 3) * 256 + t;
    out[0] = e0 * inv; out[256] = e1 * inv; out[512] = e2 * inv;
}

__global__ __launch_bounds__(256)
void k_fuse(const float* __restrict__ f0, const float* __restrict__ f1,
            const float* __restrict__ f2_, const float* __restrict__ F2,
            unsigned short* __restrict__ FB)
{
    long gid = (long)blockIdx.x * 256 + threadIdx.x;
    if (gid >= (long)BLr * 256) return;
    int c = (int)(gid & 255);
    long rq = gid >> 8;
    int b = (int)(rq / LEN), qi = (int)(rq % LEN);
    int l, side, lvlOff; lvl_of(qi, l, side, lvlOff);
    int HW = side * side, pix = qi - lvlOff;
    const float* f = ((l == 0) ? f0 : (l == 1) ? f1 : f2_)
                     + (long)b * 768 * HW + (long)c * HW + pix;
    const float* w = F2 + ((long)(b * 3 + l) * 3) * 256 + c;
    float s = f[0] * w[0] + f[(long)256 * HW] * w[256] + f[(long)512 * HW] * w[512];
    FB[gid] = f2bf_bits(s);
}

// ---------------------------------------------------------------- group norm (in place on X)
__global__ __launch_bounds__(256)
void k_groupnorm(float* __restrict__ X, const float* __restrict__ gw,
                 const float* __restrict__ gb)
{
    __shared__ float s1[256], s2[256];
    int blk = blockIdx.x;               // (b*3+l)*32 + g
    int g = blk & 31, bl = blk >> 5, l = bl % 3, b = bl / 3;
    int side = (l == 0) ? 64 : (l == 1) ? 32 : 16;
    int lvlOff = (l == 0) ? 0 : (l == 1) ? 4096 : 5120;
    int HW = side * side, n = HW * 8;
    float sum = 0.f, sq = 0.f;
    for (int i = threadIdx.x; i < n; i += 256) {
        int p = i >> 3, j = i & 7;
        float v = X[((long)(b * LEN + lvlOff + p)) * 256 + g * 8 + j];
        sum += v; sq += v * v;
    }
    s1[threadIdx.x] = sum; s2[threadIdx.x] = sq; __syncthreads();
    for (int st = 128; st > 0; st >>= 1) {
        if (threadIdx.x < st) { s1[threadIdx.x] += s1[threadIdx.x + st];
                                s2[threadIdx.x] += s2[threadIdx.x + st]; }
        __syncthreads();
    }
    float mean = s1[0] / (float)n;
    float var  = s2[0] / (float)n - mean * mean;
    float inv  = rsqrtf(var + 1e-5f);
    for (int i = threadIdx.x; i < n; i += 256) {
        int p = i >> 3, j = i & 7, ch = g * 8 + j;
        long idx = ((long)(b * LEN + lvlOff + p)) * 256 + ch;
        X[idx] = (X[idx] - mean) * inv * gw[l * 256 + ch] + gb[l * 256 + ch];
    }
}

// ---------------------------------------------------------------- encoder helpers
__global__ __launch_bounds__(256)
void k_make_q(const float* __restrict__ X, const float* __restrict__ POS,
              unsigned short* __restrict__ QB, unsigned short* __restrict__ XB)
{
    long gid = (long)blockIdx.x * 256 + threadIdx.x;
    if (gid >= (long)BLr * 256) return;
    long posIdx = gid % ((long)LEN * 256);
    float xv = X[gid];
    XB[gid] = f2bf_bits(xv);
    QB[gid] = f2bf_bits(xv + POS[posIdx]);
}

__global__ __launch_bounds__(256)
void k_softmax12(float* __restrict__ AW)
{
    long t = (long)blockIdx.x * 256 + threadIdx.x;
    if (t >= (long)BLr * Mh) return;
    float* p = AW + t * 12;
    float mx = p[0];
#pragma unroll
    for (int i = 1; i < 12; ++i) mx = fmaxf(mx, p[i]);
    float s = 0.f;
#pragma unroll
    for (int i = 0; i < 12; ++i) { float e = expf(p[i] - mx); p[i] = e; s += e; }
    float inv = 1.f / s;
#pragma unroll
    for (int i = 0; i < 12; ++i) p[i] *= inv;
}

// deformable bilinear sampling: grid (BLr, Mh), block 32 (lane = channel d)
__global__ __launch_bounds__(32)
void k_deform(const float* __restrict__ OFF, const float* __restrict__ AW,
              const float* __restrict__ VAL, unsigned short* __restrict__ ACCB)
{
    const int row = blockIdx.x;          // b*LEN + qi
    const int m   = blockIdx.y;
    const int d   = threadIdx.x;
    const int b   = row / LEN, qi = row % LEN;
    int lq, sq, oq; lvl_of(qi, lq, sq, oq);
    const int pq = qi - oq;
    const float refx = ((float)(pq % sq) + 0.5f) / (float)sq;
    const float refy = ((float)(pq / sq) + 0.5f) / (float)sq;

    const float* offp = OFF + ((long)row * Mh + m) * 24;   // (NLVL,P,2)
    const float* awp  = AW  + ((long)row * Mh + m) * 12;

    float acc = 0.f;
    const int sides[3]  = {64, 32, 16};
    const int lvloff[3] = {0, 4096, 5120};
#pragma unroll
    for (int l = 0; l < 3; ++l) {
        const int W = sides[l];
        const float fW = (float)W;
        const long base0 = (long)(b * LEN + lvloff[l]);
#pragma unroll
        for (int p = 0; p < 4; ++p) {
            const float ox = offp[(l * 4 + p) * 2 + 0];
            const float oy = offp[(l * 4 + p) * 2 + 1];
            const float px = refx * fW + ox - 0.5f;
            const float py = refy * fW + oy - 0.5f;
            const float wgt = awp[l * 4 + p];
            const float x0f = floorf(px), y0f = floorf(py);
            const float tx = px - x0f, ty = py - y0f;
            const int x0 = (int)x0f, y0 = (int)y0f;
            const int x1 = x0 + 1, y1 = y0 + 1;
            const float w00 = (1.f - tx) * (1.f - ty), w10 = tx * (1.f - ty);
            const float w01 = (1.f - tx) * ty,         w11 = tx * ty;
            if (x0 >= 0 && x0 < W && y0 >= 0 && y0 < W)
                acc += wgt * w00 * VAL[(base0 + y0 * W + x0) * 256 + m * 32 + d];
            if (x1 >= 0 && x1 < W && y0 >= 0 && y0 < W)
                acc += wgt * w10 * VAL[(base0 + y0 * W + x1) * 256 + m * 32 + d];
            if (x0 >= 0 && x0 < W && y1 >= 0 && y1 < W)
                acc += wgt * w01 * VAL[(base0 + y1 * W + x0) * 256 + m * 32 + d];
            if (x1 >= 0 && x1 < W && y1 >= 0 && y1 < W)
                acc += wgt * w11 * VAL[(base0 + y1 * W + x1) * 256 + m * 32 + d];
        }
    }
    ACCB[(long)row * 256 + m * 32 + d] = f2bf_bits(acc);
}

// x = LayerNorm(resid + add) * w + b ; also emit bf16 copy
__global__ __launch_bounds__(256)
void k_add_layernorm(float* __restrict__ X, const float* __restrict__ add,
                     const float* __restrict__ w, const float* __restrict__ b,
                     unsigned short* __restrict__ XB)
{
    __shared__ float s1[256], s2[256];
    const long row = blockIdx.x;
    const int c = threadIdx.x;
    const long i = row * 256 + c;
    float v = X[i] + add[i];
    s1[c] = v; s2[c] = v * v; __syncthreads();
    for (int st = 128; st > 0; st >>= 1) {
        if (c < st) { s1[c] += s1[c + st]; s2[c] += s2[c + st]; }
        __syncthreads();
    }
    float mean = s1[0] * (1.f / 256.f);
    float var  = s2[0] * (1.f / 256.f) - mean * mean;
    float inv  = rsqrtf(var + 1e-5f);
    float y = (v - mean) * inv * w[c] + b[c];
    X[i]  = y;
    XB[i] = f2bf_bits(y);
}

// ---------------------------------------------------------------- head
__global__ __launch_bounds__(256)
void k_conv1(const float* __restrict__ X, const float* __restrict__ W1,
             const float* __restrict__ B1, float* __restrict__ H1)
{
    const int x = blockIdx.x * 16 + threadIdx.x;
    const int y = blockIdx.y * 16 + threadIdx.y;
    const int bo = blockIdx.z, b = bo >> 6, o = bo & 63;
    float acc = B1[o];
    for (int dy = 0; dy < 3; ++dy) {
        const int yy = y + dy - 1;
        if (yy < 0 || yy >= 64) continue;
        for (int dx = 0; dx < 3; ++dx) {
            const int xx = x + dx - 1;
            if (xx < 0 || xx >= 64) continue;
            const float* xp = X + ((long)b * LEN + yy * 64 + xx) * 256;
            const float* wp = W1 + (long)o * 256 * 9 + dy * 3 + dx;
            for (int c = 0; c < 256; ++c) acc += xp[c] * wp[c * 9];
        }
    }
    H1[((long)(b * 64 + o)) * 4096 + y * 64 + x] = fmaxf(acc, 0.f);
}

__global__ __launch_bounds__(256)
void k_conv2(const float* __restrict__ H1, const float* __restrict__ W2,
             const float* __restrict__ B2, float* __restrict__ PRED)
{
    int idx = blockIdx.x * 256 + threadIdx.x;      // 8*2*64*64
    if (idx >= 65536) return;
    int x = idx & 63, y = (idx >> 6) & 63, o = (idx >> 12) & 1, b = idx >> 13;
    float acc = B2[o];
    for (int dy = 0; dy < 3; ++dy) {
        int yy = y + dy - 1; if (yy < 0 || yy >= 64) continue;
        for (int dx = 0; dx < 3; ++dx) {
            int xx = x + dx - 1; if (xx < 0 || xx >= 64) continue;
            for (int c = 0; c < 64; ++c)
                acc += H1[((long)(b * 64 + c)) * 4096 + yy * 64 + xx]
                     * W2[((o * 64 + c) * 3 + dy) * 3 + dx];
        }
    }
    PRED[((long)(b * 2 + o)) * 4096 + y * 64 + x] = acc;
}

__global__ __launch_bounds__(256)
void k_upsample(const float* __restrict__ PRED, float* __restrict__ OUT)
{
    int idx = blockIdx.x * 256 + threadIdx.x;      // 8*2*256*256
    if (idx >= 8 * 2 * 256 * 256) return;
    int X_ = idx & 255, Y = (idx >> 8) & 255, o = (idx >> 16) & 1, b = idx >> 17;
    float sy = ((float)Y + 0.5f) * 0.25f - 0.5f;
    float sx = ((float)X_ + 0.5f) * 0.25f - 0.5f;
    float y0f = floorf(sy), x0f = floorf(sx);
    float ty = sy - y0f, tx = sx - x0f;
    int y0 = min(max((int)y0f, 0), 63), y1 = min(y0 + 1, 63);
    int x0 = min(max((int)x0f, 0), 63), x1 = min(x0 + 1, 63);
    const float* p = PRED + ((long)(b * 2 + o)) * 4096;
    float v = (1.f - ty) * ((1.f - tx) * p[y0 * 64 + x0] + tx * p[y0 * 64 + x1])
            +        ty  * ((1.f - tx) * p[y1 * 64 + x0] + tx * p[y1 * 64 + x1]);
    OUT[idx] = v;
}

// ---------------------------------------------------------------- launcher
extern "C" void kernel_launch(void* const* d_in, const int* in_sizes, int n_in,
                              void* d_out, int out_size, void* d_ws, size_t ws_size,
                              hipStream_t stream)
{
    (void)in_sizes; (void)n_in; (void)out_size; (void)ws_size;
    const float* feat_l0   = (const float*)d_in[0];
    const float* feat_l1   = (const float*)d_in[1];
    const float* feat_l2   = (const float*)d_in[2];
    const float* tfl_fc1_w = (const float*)d_in[3];
    const float* tfl_fc1_b = (const float*)d_in[4];
    const float* tfl_fc2_w = (const float*)d_in[5];
    const float* tfl_fc2_b = (const float*)d_in[6];
    const float* proj_w    = (const float*)d_in[7];
    const float* proj_b    = (const float*)d_in[8];
    const float* gn_w      = (const float*)d_in[9];
    const float* gn_b      = (const float*)d_in[10];
    const float* level_emb = (const float*)d_in[11];
    const float* off_w     = (const float*)d_in[12];
    const float* off_b     = (const float*)d_in[13];
    const float* aw_w      = (const float*)d_in[14];
    const float* aw_b      = (const float*)d_in[15];
    const float* vp_w      = (const float*)d_in[16];
    const float* vp_b      = (const float*)d_in[17];
    const float* op_w      = (const float*)d_in[18];
    const float* op_b      = (const float*)d_in[19];
    const float* n1_w      = (const float*)d_in[20];
    const float* n1_b      = (const float*)d_in[21];
    const float* l1_w      = (const float*)d_in[22];
    const float* l1_b      = (const float*)d_in[23];
    const float* l2_w      = (const float*)d_in[24];
    const float* l2_b      = (const float*)d_in[25];
    const float* n2_w      = (const float*)d_in[26];
    const float* n2_b      = (const float*)d_in[27];
    const float* head1_w   = (const float*)d_in[28];
    const float* head1_b   = (const float*)d_in[29];
    const float* head2_w   = (const float*)d_in[30];
    const float* head2_b   = (const float*)d_in[31];
    float* OUT = (float*)d_out;

    // ---- workspace layout (256B aligned)
    char* ws = (char*)d_ws;
    size_t o = 0;
    auto alloc = [&](size_t bytes) { size_t r = o; o += (bytes + 255) & ~(size_t)255; return r; };
    const size_t BLs = (size_t)BLr;
    size_t oX    = alloc(BLs * 256 * 4);
    size_t oPOS  = alloc((size_t)LEN * 256 * 4);
    size_t oXB   = alloc(BLs * 256 * 2);
    size_t oQB   = alloc(BLs * 256 * 2);        // also fused bf16 pre-encoder
    size_t oOFF  = alloc(BLs * 192 * 4);        // region also aliased by HIDB / H1
    size_t oAW   = alloc(BLs * 96 * 4);         // aliased by PRED post-encoder
    size_t oVAL  = alloc(BLs * 256 * 4);
    size_t oACCB = alloc(BLs * 256 * 2);
    size_t oATTN = alloc(BLs * 256 * 4);
    size_t oSQ   = alloc((size_t)Bn * 3 * 768 * 4);
    size_t oF2   = alloc((size_t)Bn * 3 * 768 * 4);
    size_t oWPJ  = alloc((size_t)3 * 65536 * 2);
    size_t oWOF  = alloc((size_t)6 * 256 * 192 * 2);
    size_t oWAW  = alloc((size_t)6 * 256 * 96 * 2);
    size_t oWVP  = alloc((size_t)6 * 65536 * 2);
    size_t oWOP  = alloc((size_t)6 * 65536 * 2);
    size_t oWL1  = alloc((size_t)6 * 262144 * 2);
    size_t oWL2  = alloc((size_t)6 * 262144 * 2);
    size_t oHIDB = oOFF;                        // BL*1024*2 fits inside OFF+AW+VAL
    size_t oH1   = oOFF;                        // head temps (after encoder)
    size_t oPRED = oAW;

    float*          X    = (float*)(ws + oX);
    float*          POS  = (float*)(ws + oPOS);
    unsigned short* XB   = (unsigned short*)(ws + oXB);
    unsigned short* QB   = (unsigned short*)(ws + oQB);
    float*          OFF  = (float*)(ws + oOFF);
    float*          AW   = (float*)(ws + oAW);
    float*          VAL  = (float*)(ws + oVAL);
    unsigned short* ACCB = (unsigned short*)(ws + oACCB);
    float*          ATTN = (float*)(ws + oATTN);
    float*          SQ   = (float*)(ws + oSQ);
    float*          F2   = (float*)(ws + oF2);
    unsigned short* WPJ  = (unsigned short*)(ws + oWPJ);
    unsigned short* WOF  = (unsigned short*)(ws + oWOF);
    unsigned short* WAWt = (unsigned short*)(ws + oWAW);
    unsigned short* WVP  = (unsigned short*)(ws + oWVP);
    unsigned short* WOP  = (unsigned short*)(ws + oWOP);
    unsigned short* WL1  = (unsigned short*)(ws + oWL1);
    unsigned short* WL2  = (unsigned short*)(ws + oWL2);
    unsigned short* HIDB = (unsigned short*)(ws + oHIDB);
    float*          H1   = (float*)(ws + oH1);
    float*          PRED = (float*)(ws + oPRED);

    auto gblk = [](long n) { return (unsigned)((n + 255) / 256); };
    auto gemm = [&](const unsigned short* A, const unsigned short* Bt, const float* bias,
                    float* Cf, unsigned short* Cb, int M, int N, int K,
                    long sAz, long sCz, int Z, int relu) {
        dim3 blk(32, 4, 1);
        dim3 grd(M / 16, (N / 16 + 3) / 4, Z);
        gemm_bf16<<<grd, blk, 0, stream>>>(A, Bt, bias, Cf, Cb, M, N, K, sAz, sCz, relu);
    };

    // ---- weights -> bf16 (transposed to column-major where needed)
    k_w_transpose<<<gblk(6L * 256 * 192),  256, 0, stream>>>(off_w, WOF, 256, 192,  6L * 256 * 192);
    k_w_transpose<<<gblk(6L * 256 * 96),   256, 0, stream>>>(aw_w,  WAWt, 256, 96,  6L * 256 * 96);
    k_w_transpose<<<gblk(6L * 65536),      256, 0, stream>>>(vp_w,  WVP, 256, 256,  6L * 65536);
    k_w_transpose<<<gblk(6L * 65536),      256, 0, stream>>>(op_w,  WOP, 256, 256,  6L * 65536);
    k_w_transpose<<<gblk(6L * 262144),     256, 0, stream>>>(l1_w,  WL1, 256, 1024, 6L * 262144);
    k_w_transpose<<<gblk(6L * 262144),     256, 0, stream>>>(l2_w,  WL2, 1024, 256, 6L * 262144);
    // proj_w is (o,c): already column-major for C[pixel,o] = sum_c fused*W[o,c]
    k_w_convert<<<gblk(3L * 65536), 256, 0, stream>>>(proj_w, WPJ, 3L * 65536);

    // ---- position embeddings
    k_pos<<<gblk((long)LEN * 256), 256, 0, stream>>>(level_emb, POS);

    // ---- time fusion + projection + group norm
    k_squeeze<<<Bn * 3 * 768, 256, 0, stream>>>(feat_l0, feat_l1, feat_l2, SQ);
    k_tfl<<<Bn * 3, 256, 0, stream>>>(SQ, tfl_fc1_w, tfl_fc1_b, tfl_fc2_w, tfl_fc2_b, F2);
    k_fuse<<<gblk((long)BLr * 256), 256, 0, stream>>>(feat_l0, feat_l1, feat_l2, F2, QB);
    const int sides[3] = {64, 32, 16};
    const int loffs[3] = {0, 4096, 5120};
    for (int l = 0; l < 3; ++l) {
        int HW = sides[l] * sides[l];
        gemm(QB + (long)loffs[l] * 256, WPJ + (long)l * 65536, proj_b + l * 256,
             X + (long)loffs[l] * 256, nullptr, HW, 256, 256,
             (long)LEN * 256, (long)LEN * 256, Bn, 0);
    }
    k_groupnorm<<<Bn * 3 * 32, 256, 0, stream>>>(X, gn_w, gn_b);

    // ---- encoder layers
    for (int li = 0; li < NLAY; ++li) {
        k_make_q<<<gblk((long)BLr * 256), 256, 0, stream>>>(X, POS, QB, XB);
        gemm(QB, WOF + (long)li * 256 * 192, off_b + li * 192,
             OFF, nullptr, BLr, 192, 256, 0, 0, 1, 0);
        gemm(QB, WAWt + (long)li * 256 * 96, aw_b + li * 96,
             AW, nullptr, BLr, 96, 256, 0, 0, 1, 0);
        gemm(XB, WVP + (long)li * 65536, vp_b + li * 256,
             VAL, nullptr, BLr, 256, 256, 0, 0, 1, 0);
        k_softmax12<<<gblk((long)BLr * Mh), 256, 0, stream>>>(AW);
        {
            dim3 grd(BLr, Mh, 1);
            k_deform<<<grd, 32, 0, stream>>>(OFF, AW, VAL, ACCB);
        }
        gemm(ACCB, WOP + (long)li * 65536, op_b + li * 256,
             ATTN, nullptr, BLr, 256, 256, 0, 0, 1, 0);
        k_add_layernorm<<<BLr, 256, 0, stream>>>(X, ATTN, n1_w + li * 256, n1_b + li * 256, XB);
        gemm(XB, WL1 + (long)li * 262144, l1_b + li * 1024,
             nullptr, HIDB, BLr, 1024, 256, 0, 0, 1, 1);
        gemm(HIDB, WL2 + (long)li * 262144, l2_b + li * 256,
             ATTN, nullptr, BLr, 256, 1024, 0, 0, 1, 0);
        k_add_layernorm<<<BLr, 256, 0, stream>>>(X, ATTN, n2_w + li * 256, n2_b + li * 256, XB);
    }

    // ---- segmentation head
    {
        dim3 blk(16, 16, 1);
        dim3 grd(4, 4, Bn * 64);
        k_conv1<<<grd, blk, 0, stream>>>(X, head1_w, head1_b, H1);
    }
    k_conv2<<<gblk(65536), 256, 0, stream>>>(H1, head2_w, head2_b, PRED);
    k_upsample<<<gblk(8L * 2 * 256 * 256), 256, 0, stream>>>(PRED, OUT);
}
